// Stacked_LSTM_55336358642684
// MI455X (gfx1250) — matrix operand order repaired and checked by reading, not compiled, not run on hardware
//
#include <hip/hip_runtime.h>
#include <hip/hip_bf16.h>

typedef __attribute__((ext_vector_type(16))) _Float16 v16h;
typedef __attribute__((ext_vector_type(8)))  _Float16 v8h;
typedef __attribute__((ext_vector_type(8)))  float    v8f;

#define B_TOT    512
#define T_LEN    1024
#define H        128
#define BT       16          // batch rows per workgroup (one WMMA M-tile)
#define NTHREADS 512         // 16 wave32s -> 4 waves/SIMD -> 256 VGPR budget/wave
#define GPAD     516         // padded gate-row stride (floats)
#define XPAD     1032        // padded input-row stride (floats)
#define WSTR     136         // padded W_hh1 LDS row stride in halfs (272B, 16B-aligned)

__device__ __forceinline__ float sigm(float x) {
  // 1 / (1 + e^-x)  -> v_exp_f32 + v_rcp_f32 (TRANS, co-executes with WMMA)
  return __builtin_amdgcn_rcpf(1.0f + __expf(-x));
}
__device__ __forceinline__ float tanha(float x) {
  return 2.0f * sigm(2.0f * x) - 1.0f;
}

__device__ __forceinline__ v8f wmma_f16(v16h a, v16h b, v8f c) {
  // D = A(16x32 f16) * B(32x16 f16) + C(16x16 f32)
  return __builtin_amdgcn_wmma_f32_16x16x32_f16(
      false, a, false, b, (short)0, c, false, false);
}

// A-fragment (16x32 f16, k-step s) from row-major [16][128] f16 LDS tile.
// Lane L (M = L&15) holds K = 32s + (L>=16?16:0) + 0..15 -> 2x ds_load_b128.
__device__ __forceinline__ v16h load_a_frag(const _Float16* hbase, int lane, int s) {
  const int m  = lane & 15;
  const int kb = 32 * s + ((lane >= 16) ? 16 : 0);
  union { v16h v; v8h p[2]; } u;
  u.p[0] = *reinterpret_cast<const v8h*>(hbase + m * H + kb);
  u.p[1] = *reinterpret_cast<const v8h*>(hbase + m * H + kb + 8);
  return u.v;
}

// B-fragment (32x16 f16, k-step s) for gate column n from the padded LDS
// f16 weight copy. Lane L holds column N=L&15, K = 32s + (L>=16?16:0) + 0..15:
// 16 contiguous halfs = 32 contiguous bytes -> 2x ds_load_b128.
__device__ __forceinline__ v16h load_b_frag(const _Float16* wbase, int n, int lane, int s) {
  const int kb = 32 * s + ((lane >= 16) ? 16 : 0);
  const _Float16* p = wbase + n * WSTR + kb;
  union { v16h v; v8h q[2]; } u;
  u.q[0] = *reinterpret_cast<const v8h*>(p);
  u.q[1] = *reinterpret_cast<const v8h*>(p + 8);
  return u.v;
}

__global__ __launch_bounds__(NTHREADS, 1)
void lstm_stack_kernel(const float* __restrict__ xin,
                       const float* __restrict__ W_ih1, const float* __restrict__ W_hh1,
                       const float* __restrict__ b_ih1, const float* __restrict__ b_hh1,
                       const float* __restrict__ W_ih2, const float* __restrict__ W_hh2,
                       const float* __restrict__ b_ih2, const float* __restrict__ b_hh2,
                       const float* __restrict__ W_out, const float* __restrict__ b_out,
                       float* __restrict__ out)
{
  __shared__ float xall[BT][XPAD];                   // staged input rows (f32)
  __shared__ float glds[BT][GPAD];                   // gate exchange buffer (f32)
  __shared__ alignas(16) _Float16 whh1[512][WSTR];   // W_hh1 as f16, padded rows
  __shared__ alignas(16) _Float16 hA1[BT][H];        // h1 state, row-major f16
  __shared__ alignas(16) _Float16 hA2[BT][H];        // h2 state, row-major f16
  __shared__ float part[BT][4];                      // output-projection partials

  const int tid   = threadIdx.x;
  const int lane  = tid & 31;
  const int w     = tid >> 5;            // wave id 0..15
  const int bbase = blockIdx.x * BT;
  const int ncb   = w * 32;              // this wave's gate-column base (2 N-tiles)
  const int ncol  = lane & 15;           // C/D: N = lane&15
  const int mlo   = (lane >= 16) ? 8 : 0;// C/D: M = r + (lane>=16 ? 8 : 0)

  // ---- stage input rows + W_hh1(f16) into LDS, zero h state ----
  for (int idx = tid; idx < BT * T_LEN; idx += NTHREADS) {
    const int b = idx >> 10, t = idx & 1023;
    xall[b][t] = xin[(bbase + b) * T_LEN + t];
  }
  for (int idx = tid; idx < 512 * H; idx += NTHREADS) {
    const int nn = idx >> 7, kk = idx & 127;
    whh1[nn][kk] = (_Float16)W_hh1[idx];
  }
  for (int idx = tid; idx < BT * H; idx += NTHREADS) {
    hA1[idx >> 7][idx & 127] = (_Float16)0.0f;
    hA2[idx >> 7][idx & 127] = (_Float16)0.0f;
  }

  // ---- one-time: layer-2 weights f32->f16 into register-resident B fragments ----
  // 16 fragments * 8 VGPRs = 128 VGPRs per lane (fits the 256-VGPR budget).
  v16h Bih2[2][4], Bhh2[2][4];
  float wx[2], bias1[2], bias2[2];
  int   ntile[2];
#pragma unroll
  for (int tIdx = 0; tIdx < 2; ++tIdx) {
    const int n = ncb + tIdx * 16 + ncol;
    ntile[tIdx] = n;
    wx[tIdx]    = W_ih1[n];                  // W_ih1 is [512,1]
    bias1[tIdx] = b_ih1[n] + b_hh1[n];
    bias2[tIdx] = b_ih2[n] + b_hh2[n];
#pragma unroll
    for (int s = 0; s < 4; ++s) {
      const int kb = 32 * s + ((lane >= 16) ? 16 : 0);
      const float* p2 = W_ih2 + n * H + kb;
      const float* p3 = W_hh2 + n * H + kb;
      v16h f2, f3;
#pragma unroll
      for (int h = 0; h < 16; ++h) {
        f2[h] = (_Float16)p2[h];
        f3[h] = (_Float16)p3[h];
      }
      Bih2[tIdx][s] = f2; Bhh2[tIdx][s] = f3;
    }
  }

  const int   jj   = tid & 127;    // hidden index for elementwise phase
  const int   bq   = tid >> 7;     // base batch row (cells b = bq + 4k, k=0..3)
  const float wout = W_out[jj];
  const float bOut = b_out[0];
  float c1r[4] = {0.f, 0.f, 0.f, 0.f};
  float c2r[4] = {0.f, 0.f, 0.f, 0.f};

  __syncthreads();

  for (int t = 0; t < T_LEN; ++t) {
    // ====== layer 1: gates1 = h1 @ W_hh1^T, B streamed from LDS f16 copy ======
    {
      v8f d0 = {}, d1 = {};
#pragma unroll
      for (int s = 0; s < 4; ++s) {
        v16h a  = load_a_frag(&hA1[0][0], lane, s);
        v16h b0 = load_b_frag(&whh1[0][0], ntile[0], lane, s);
        v16h b1 = load_b_frag(&whh1[0][0], ntile[1], lane, s);
        d0 = wmma_f16(a, b0, d0);
        d1 = wmma_f16(a, b1, d1);
      }
#pragma unroll
      for (int r = 0; r < 8; ++r) {
        const float xv = xall[mlo + r][t];
        glds[mlo + r][ncb + ncol]      = d0[r] + xv * wx[0] + bias1[0];
        glds[mlo + r][ncb + 16 + ncol] = d1[r] + xv * wx[1] + bias1[1];
      }
    }
    __syncthreads();

    // ====== layer 1 elementwise: c1,h1 update (c1 in regs) ======
#pragma unroll
    for (int k = 0; k < 4; ++k) {
      const int b = bq + 4 * k;
      const float gi = glds[b][jj];
      const float gf = glds[b][H + jj];
      const float gg = glds[b][2 * H + jj];
      const float go = glds[b][3 * H + jj];
      const float c  = sigm(gf) * c1r[k] + sigm(gi) * tanha(gg);
      c1r[k] = c;
      hA1[b][jj] = (_Float16)(sigm(go) * tanha(c));
    }
    __syncthreads();

    // ====== layer 2: gates2 = h1 @ W_ih2^T + h2 @ W_hh2^T (B in registers) ======
    {
      v8f d0 = {}, d1 = {};
#pragma unroll
      for (int s = 0; s < 4; ++s) {
        v16h a = load_a_frag(&hA1[0][0], lane, s);
        d0 = wmma_f16(a, Bih2[0][s], d0);
        d1 = wmma_f16(a, Bih2[1][s], d1);
      }
#pragma unroll
      for (int s = 0; s < 4; ++s) {
        v16h a = load_a_frag(&hA2[0][0], lane, s);
        d0 = wmma_f16(a, Bhh2[0][s], d0);
        d1 = wmma_f16(a, Bhh2[1][s], d1);
      }
#pragma unroll
      for (int r = 0; r < 8; ++r) {
        glds[mlo + r][ncb + ncol]      = d0[r] + bias2[0];
        glds[mlo + r][ncb + 16 + ncol] = d1[r] + bias2[1];
      }
    }
    __syncthreads();

    // ====== layer 2 elementwise + output projection (deterministic reduction) ======
#pragma unroll
    for (int k = 0; k < 4; ++k) {
      const int b = bq + 4 * k;        // all 32 lanes of a wave share b
      const float gi = glds[b][jj];
      const float gf = glds[b][H + jj];
      const float gg = glds[b][2 * H + jj];
      const float go = glds[b][3 * H + jj];
      const float c  = sigm(gf) * c2r[k] + sigm(gi) * tanha(gg);
      c2r[k] = c;
      const float h  = sigm(go) * tanha(c);
      hA2[b][jj] = (_Float16)h;
      float p = h * wout;
#pragma unroll
      for (int off = 16; off >= 1; off >>= 1)
        p += __shfl_xor(p, off, 32);   // wave32 butterfly reduce
      if (lane == 0) part[b][w & 3] = p;
    }
    __syncthreads();

    if (tid < BT)
      out[(bbase + tid) * T_LEN + t] =
          part[tid][0] + part[tid][1] + part[tid][2] + part[tid][3] + bOut;
    __syncthreads();
  }
}

extern "C" void kernel_launch(void* const* d_in, const int* in_sizes, int n_in,
                              void* d_out, int out_size, void* d_ws, size_t ws_size,
                              hipStream_t stream) {
  (void)in_sizes; (void)n_in; (void)out_size; (void)d_ws; (void)ws_size;
  const float* xin  = (const float*)d_in[0];
  const float* Wih1 = (const float*)d_in[1];
  const float* Whh1 = (const float*)d_in[2];
  const float* bih1 = (const float*)d_in[3];
  const float* bhh1 = (const float*)d_in[4];
  const float* Wih2 = (const float*)d_in[5];
  const float* Whh2 = (const float*)d_in[6];
  const float* bih2 = (const float*)d_in[7];
  const float* bhh2 = (const float*)d_in[8];
  const float* Wout = (const float*)d_in[9];
  const float* bout = (const float*)d_in[10];
  float* out = (float*)d_out;

  lstm_stack_kernel<<<dim3(B_TOT / BT), dim3(NTHREADS), 0, stream>>>(
      xin, Wih1, Whh1, bih1, bhh1, Wih2, Whh2, bih2, bhh2, Wout, bout, out);
}
